// AutoGenReview_21114059227695
// MI455X (gfx1250) — compile-verified, matrix operand order checked by reading, not run on hardware
//
#include <hip/hip_runtime.h>

// ---------------- constants ----------------
#define BB 128
#define TT 32
#define DD 4
#define NI 50000
#define UD 64
#define IDM 64
#define L0 128
#define L1 64
#define VV 32000
#define VW 128
#define HH 256
#define LSTM_IN 256
#define G4H 1024   // 4*H

// ---------------- WMMA fragment types ----------------
typedef __attribute__((ext_vector_type(16))) __bf16    v16bf;
typedef __attribute__((ext_vector_type(8)))  float     v8f;
typedef __attribute__((ext_vector_type(4)))  unsigned  v4u;

union Frag16 { v16bf v; v4u q[2]; };

// Load a 16-element bf16 fragment as two 16B chunks (ISA bf16 A/B lane layout:
// lane holds K = {k0..k0+7, k0+16..k0+23} where k0 = (lane>>4)*8).
__device__ inline v16bf load_frag16(const unsigned short* p0,
                                    const unsigned short* p1) {
  Frag16 f;
  f.q[0] = *reinterpret_cast<const v4u*>(p0);
  f.q[1] = *reinterpret_cast<const v4u*>(p1);
  return f.v;
}

__device__ inline unsigned short f2bf(float f) {
  unsigned u = __float_as_uint(f);
  unsigned r = u + 0x7FFFu + ((u >> 16) & 1u);   // round-to-nearest-even
  return (unsigned short)(r >> 16);
}

__device__ inline float sigm(float x) { return 1.f / (1.f + expf(-x)); }

// ---------------- misc init: bias fold, state zero, pack offsets ----------------
__global__ void init_kernel(const float* __restrict__ b_ih,
                            const float* __restrict__ b_hh,
                            float* __restrict__ bias,
                            const int* __restrict__ length,
                            int* __restrict__ offs,
                            unsigned short* __restrict__ h0,
                            float* __restrict__ cst) {
  int i = blockIdx.x * blockDim.x + threadIdx.x;
  int stride = gridDim.x * blockDim.x;
  for (int k = i; k < G4H; k += stride) bias[k] = b_ih[k] + b_hh[k];
  for (int k = i; k < BB * HH; k += stride) { h0[k] = 0; cst[k] = 0.f; }
  if (i == 0) {
    int off = 0;
    for (int t = 0; t < TT; ++t) {
      offs[t] = off;
      int cnt = 0;
      for (int b = 0; b < BB; ++b) cnt += (length[b] > t) ? 1 : 0;
      off += cnt;
    }
    offs[TT] = off;
  }
}

// ---------------- f32 -> bf16 weight conversion ----------------
__global__ void cvt_kernel(const float* __restrict__ src,
                           unsigned short* __restrict__ dst, int n) {
  int i = blockIdx.x * blockDim.x + threadIdx.x;
  if (i < n) dst[i] = f2bf(src[i]);
}

// ---------------- embeddings + domain-routed expert MLP + rating ----------------
__global__ __launch_bounds__(128) void expert_kernel(
    const int* __restrict__ u_idx, const int* __restrict__ i_idx,
    const int* __restrict__ d_idx,
    const float* __restrict__ emb_u, const float* __restrict__ emb_i,
    const float* __restrict__ w0, const float* __restrict__ b0,
    const float* __restrict__ w1, const float* __restrict__ b1,
    const float* __restrict__ aw, const float* __restrict__ ab,
    float* __restrict__ rating, float* __restrict__ uv_ws) {
  const int b = blockIdx.x, tid = threadIdx.x;
  __shared__ float uv[L0], hh0[L0], hh1[L1], red[L1];
  const int d = d_idx[b];
  float v;
  if (tid < UD) v = emb_u[(size_t)u_idx[b] * UD + tid];
  else          v = emb_i[((size_t)d * NI + i_idx[b]) * IDM + (tid - UD)];
  uv[tid] = v;
  uv_ws[b * L0 + tid] = v;
  __syncthreads();
  // layer 0: [128] x [128,128]
  float s = b0[d * L0 + tid];
  const float* wp = w0 + (size_t)d * L0 * L0 + tid;
  for (int k = 0; k < L0; ++k) s += uv[k] * wp[(size_t)k * L0];
  hh0[tid] = fmaxf(s, 0.f);
  __syncthreads();
  // layer 1: [128] x [128,64], affine partials
  if (tid < L1) {
    float s1 = b1[d * L1 + tid];
    const float* wp1 = w1 + (size_t)d * L0 * L1 + tid;
    for (int k = 0; k < L0; ++k) s1 += hh0[k] * wp1[(size_t)k * L1];
    float h = fmaxf(s1, 0.f);
    hh1[tid] = h;
    red[tid] = h * aw[tid];
  }
  __syncthreads();
  for (int off = L1 / 2; off > 0; off >>= 1) {
    if (tid < off) red[tid] += red[tid + off];
    __syncthreads();
  }
  if (tid == 0) rating[b] = 5.f * sigm(red[0] + ab[0]);
}

// ---------------- build bf16 LSTM input x[T,B,256] = [uv || word_emb] ----------------
__global__ __launch_bounds__(128) void build_x_kernel(
    const float* __restrict__ uv_ws, const float* __restrict__ word_emb,
    const int* __restrict__ review, unsigned short* __restrict__ X) {
  const int blk = blockIdx.x;          // t*B + b
  const int t = blk >> 7;              // / 128
  const int b = blk & 127;
  const int c = threadIdx.x;           // 0..127
  const size_t base = (size_t)blk * LSTM_IN;
  X[base + c] = f2bf(uv_ws[b * L0 + c]);
  const int w = review[b * TT + t];    // review is [B,T]
  X[base + VW + c] = f2bf(word_emb[(size_t)w * VW + c]);
}

// ---------------- one LSTM step: 4 waves = 4 gates, WMMA K=256 over x and h ----------------
__global__ __launch_bounds__(128) void lstm_step_kernel(
    const unsigned short* __restrict__ Xt,     // [B,256] bf16 (slice at t)
    const unsigned short* __restrict__ Hprev,  // [B,256] bf16
    const unsigned short* __restrict__ WihB,   // [1024,256] bf16
    const unsigned short* __restrict__ WhhB,   // [1024,256] bf16
    const float* __restrict__ bias,            // [1024]
    float* __restrict__ Cst,                   // [B,256] f32 (in-place)
    unsigned short* __restrict__ Hout)         // [B,256] bf16 (hs slice at t)
{
  const int tid  = threadIdx.x;
  const int wave = tid >> 5;            // gate: 0=i 1=f 2=g 3=o
  const int lane = tid & 31;
  const int ln   = lane & 15;           // A row / B col within tile
  const int kh   = (lane >> 4) * 8;     // K-half select
  const int hc   = blockIdx.x;          // H col tile (16 of them)
  const int rb   = blockIdx.y * 16;     // batch row base
  const int gcol = wave * HH + hc * 16; // gate column base in [0,1024)

  v8f acc = {};
  const unsigned short* arow = Xt   + (size_t)(rb + ln) * LSTM_IN;
  const unsigned short* brow = WihB + (size_t)(gcol + ln) * LSTM_IN;
#pragma unroll
  for (int kk = 0; kk < 8; ++kk) {
    const int k0 = kk * 32 + kh;
    v16bf a = load_frag16(arow + k0, arow + k0 + 16);
    v16bf b = load_frag16(brow + k0, brow + k0 + 16);
    acc = __builtin_amdgcn_wmma_f32_16x16x32_bf16(false, a, false, b,
                                                  (short)0, acc, false, false);
  }
  const unsigned short* arow2 = Hprev + (size_t)(rb + ln) * HH;
  const unsigned short* brow2 = WhhB + (size_t)(gcol + ln) * HH;
#pragma unroll
  for (int kk = 0; kk < 8; ++kk) {
    const int k0 = kk * 32 + kh;
    v16bf a = load_frag16(arow2 + k0, arow2 + k0 + 16);
    v16bf b = load_frag16(brow2 + k0, brow2 + k0 + 16);
    acc = __builtin_amdgcn_wmma_f32_16x16x32_bf16(false, a, false, b,
                                                  (short)0, acc, false, false);
  }
  const float bv = bias[gcol + ln];

  __shared__ float gl[4][16][16];
  const int mbase = (lane >> 4) * 8;    // C/D layout: lane half selects rows 0-7 / 8-15
#pragma unroll
  for (int r = 0; r < 8; ++r) gl[wave][mbase + r][ln] = acc[r] + bv;
  __syncthreads();

  // fused cell update: 256 elements, 2 per thread
#pragma unroll
  for (int j = 0; j < 2; ++j) {
    const int e = tid * 2 + j;
    const int m = e >> 4, n = e & 15;
    const float gi = gl[0][m][n], gf = gl[1][m][n];
    const float gg = gl[2][m][n], go = gl[3][m][n];
    const size_t idx = (size_t)(rb + m) * HH + (hc * 16 + n);
    const float cprev = Cst[idx];
    const float c = sigm(gf) * cprev + sigm(gi) * tanhf(gg);
    Cst[idx] = c;
    Hout[idx] = f2bf(sigm(go) * tanhf(c));
  }
}

// ---------------- packed vocab projection: [P,256] x [256,32000] ----------------
// 4 waves/block; each wave owns 4 vocab column tiles sharing one A fragment,
// so the inner K-step issues 10 b128 loads for 4 WMMAs (A reuse in registers).
__global__ __launch_bounds__(128) void out_proj_kernel(
    const unsigned short* __restrict__ Hs,    // [T,B,256] bf16
    const unsigned short* __restrict__ LinB,  // [V,256] bf16
    const float* __restrict__ lin_b,          // [V]
    const int* __restrict__ offs,             // [T+1]
    float* __restrict__ outp)                 // [P,V]
{
  const int t  = blockIdx.y >> 3;
  const int rt = blockIdx.y & 7;
  const int cnt = offs[t + 1] - offs[t];
  const int rowbase = rt * 16;
  if (rowbase >= cnt) return;               // whole tile past packed count
  const int tid  = threadIdx.x;
  const int wave = tid >> 5, lane = tid & 31;
  const int ln   = lane & 15;
  const int kh   = (lane >> 4) * 8;
  const int cb   = (blockIdx.x * 4 + wave) * 64;  // 4 tiles of 16 cols per wave

  const unsigned short* arow = Hs + ((size_t)t * BB + rowbase + ln) * HH;
  const unsigned short* br0 = LinB + (size_t)(cb +  0 + ln) * HH;
  const unsigned short* br1 = LinB + (size_t)(cb + 16 + ln) * HH;
  const unsigned short* br2 = LinB + (size_t)(cb + 32 + ln) * HH;
  const unsigned short* br3 = LinB + (size_t)(cb + 48 + ln) * HH;
  v8f acc0 = {}, acc1 = {}, acc2 = {}, acc3 = {};
#pragma unroll
  for (int kk = 0; kk < 8; ++kk) {
    const int k0 = kk * 32 + kh;
    v16bf a  = load_frag16(arow + k0, arow + k0 + 16);
    v16bf b0 = load_frag16(br0 + k0, br0 + k0 + 16);
    v16bf b1 = load_frag16(br1 + k0, br1 + k0 + 16);
    v16bf b2 = load_frag16(br2 + k0, br2 + k0 + 16);
    v16bf b3 = load_frag16(br3 + k0, br3 + k0 + 16);
    acc0 = __builtin_amdgcn_wmma_f32_16x16x32_bf16(false, a, false, b0,
                                                   (short)0, acc0, false, false);
    acc1 = __builtin_amdgcn_wmma_f32_16x16x32_bf16(false, a, false, b1,
                                                   (short)0, acc1, false, false);
    acc2 = __builtin_amdgcn_wmma_f32_16x16x32_bf16(false, a, false, b2,
                                                   (short)0, acc2, false, false);
    acc3 = __builtin_amdgcn_wmma_f32_16x16x32_bf16(false, a, false, b3,
                                                   (short)0, acc3, false, false);
  }
  const int mbase = (lane >> 4) * 8;
  const size_t obase = (size_t)offs[t];
#pragma unroll
  for (int r = 0; r < 8; ++r) {
    const int bloc = rowbase + mbase + r;
    if (bloc < cnt) {
      float* orow = outp + (obase + bloc) * (size_t)VV;
      orow[cb +  0 + ln] = acc0[r] + lin_b[cb +  0 + ln];
      orow[cb + 16 + ln] = acc1[r] + lin_b[cb + 16 + ln];
      orow[cb + 32 + ln] = acc2[r] + lin_b[cb + 32 + ln];
      orow[cb + 48 + ln] = acc3[r] + lin_b[cb + 48 + ln];
    }
  }
}

// ---------------- host side ----------------
extern "C" void kernel_launch(void* const* d_in, const int* in_sizes, int n_in,
                              void* d_out, int out_size, void* d_ws, size_t ws_size,
                              hipStream_t stream) {
  const int*   u_idx   = (const int*)  d_in[0];
  const int*   i_idx   = (const int*)  d_in[1];
  const int*   d_idx   = (const int*)  d_in[2];
  const int*   review  = (const int*)  d_in[3];
  const int*   length  = (const int*)  d_in[4];
  const float* emb_u   = (const float*)d_in[5];
  const float* emb_i   = (const float*)d_in[6];
  const float* word_w  = (const float*)d_in[7];
  const float* fc_w0   = (const float*)d_in[8];
  const float* fc_b0   = (const float*)d_in[9];
  const float* fc_w1   = (const float*)d_in[10];
  const float* fc_b1   = (const float*)d_in[11];
  const float* aff_w   = (const float*)d_in[12];
  const float* aff_b   = (const float*)d_in[13];
  const float* w_ih    = (const float*)d_in[14];
  const float* w_hh    = (const float*)d_in[15];
  const float* b_ih    = (const float*)d_in[16];
  const float* b_hh    = (const float*)d_in[17];
  const float* lin_w   = (const float*)d_in[18];
  const float* lin_b   = (const float*)d_in[19];

  float* rating = (float*)d_out;          // [128]
  float* outp   = (float*)d_out + BB;     // [P, V]

  // workspace carve-up (16B aligned slabs)
  char* ws = (char*)d_ws;
  size_t off = 0;
  auto take = [&](size_t bytes) { char* p = ws + off; off = (off + bytes + 15) & ~(size_t)15; return p; };
  float*          uv_ws  = (float*)         take((size_t)BB * L0 * 4);
  unsigned short* X      = (unsigned short*)take((size_t)TT * BB * LSTM_IN * 2);
  unsigned short* wih_b  = (unsigned short*)take((size_t)G4H * LSTM_IN * 2);
  unsigned short* whh_b  = (unsigned short*)take((size_t)G4H * HH * 2);
  unsigned short* lin_bf = (unsigned short*)take((size_t)VV * HH * 2);
  float*          bias   = (float*)         take((size_t)G4H * 4);
  unsigned short* hs     = (unsigned short*)take((size_t)TT * BB * HH * 2);
  unsigned short* h0     = (unsigned short*)take((size_t)BB * HH * 2);
  float*          cst    = (float*)         take((size_t)BB * HH * 4);
  int*            offs   = (int*)           take((size_t)(TT + 1) * 4);

  // 1. init: bias fold, state zero, pack offsets
  init_kernel<<<64, 256, 0, stream>>>(b_ih, b_hh, bias, length, offs, h0, cst);

  // 2. weight conversions to bf16
  {
    int n1 = G4H * LSTM_IN;
    cvt_kernel<<<(n1 + 255) / 256, 256, 0, stream>>>(w_ih, wih_b, n1);
    int n2 = G4H * HH;
    cvt_kernel<<<(n2 + 255) / 256, 256, 0, stream>>>(w_hh, whh_b, n2);
    int n3 = VV * HH;
    cvt_kernel<<<(n3 + 255) / 256, 256, 0, stream>>>(lin_w, lin_bf, n3);
  }

  // 3. embeddings + expert MLP + rating (also stages uv)
  expert_kernel<<<BB, 128, 0, stream>>>(u_idx, i_idx, d_idx, emb_u, emb_i,
                                        fc_w0, fc_b0, fc_w1, fc_b1,
                                        aff_w, aff_b, rating, uv_ws);

  // 4. build bf16 LSTM input
  build_x_kernel<<<TT * BB, 128, 0, stream>>>(uv_ws, word_w, review, X);

  // 5. LSTM: 32 sequential WMMA steps
  for (int t = 0; t < TT; ++t) {
    const unsigned short* Xt = X + (size_t)t * BB * LSTM_IN;
    const unsigned short* Hp = (t == 0) ? h0 : hs + (size_t)(t - 1) * BB * HH;
    unsigned short* Ho = hs + (size_t)t * BB * HH;
    lstm_step_kernel<<<dim3(HH / 16, BB / 16), 128, 0, stream>>>(
        Xt, Hp, wih_b, whh_b, bias, cst, Ho);
  }

  // 6. packed vocab projection: 125 x 256 blocks, 128 threads (4 waves x 4 tiles)
  out_proj_kernel<<<dim3(VV / 256, TT * 8), 128, 0, stream>>>(
      hs, lin_bf, lin_b, offs, outp);
}
